// MotionPredictionNN_87746181857929
// MI455X (gfx1250) — matrix operand order repaired and checked by reading, not compiled, not run on hardware
//
#include <hip/hip_runtime.h>

// CDNA5 / gfx1250: wave32, v_wmma_f32_16x16x32_bf16 (f32 accum).
typedef __attribute__((ext_vector_type(16))) __bf16 v16bf;
typedef __attribute__((ext_vector_type(8)))  __bf16 v8bf;
typedef __attribute__((ext_vector_type(8)))  float  v8f;
typedef __attribute__((ext_vector_type(4)))  float  f32x4;

#define MOE_BATCH 128
#define MOE_NEXP  8

// ---------------------------------------------------------------------------
// Streaming f32 -> bf16 convert (8 elements / thread, b128 loads).
// ---------------------------------------------------------------------------
__global__ __launch_bounds__(256)
void f32_to_bf16_kernel(const float* __restrict__ src,
                        __bf16* __restrict__ dst, int n) {
    int i = (blockIdx.x * blockDim.x + threadIdx.x) * 8;
    if (i + 7 < n) {
        f32x4 v0 = *(const f32x4*)(src + i);
        f32x4 v1 = *(const f32x4*)(src + i + 4);
        v8bf o;
#pragma unroll
        for (int j = 0; j < 4; ++j) {
            o[j]     = (__bf16)v0[j];
            o[4 + j] = (__bf16)v1[j];
        }
        *(v8bf*)(dst + i) = o;
    }
}

// ---------------------------------------------------------------------------
// Blended-MoE layer, expert-per-wave decomposition:
//   out[m,n] = act( sum_e BC[m,e] * ( (A W_e^T)[m,n] + EB_e[n] ) )
//
// Block = 8 waves; wave e computes the 16x16 (mtile,ntile) output tile for
// expert e with a single WMMA accumulator chain (2-deep to break the C RAW
// chain), so the compiler can pipeline B loads across K-steps. Grid =
// (N/16, M/16) blocks -> 8x more waves in flight than the expert-in-registers
// version. The expert blend is an LDS reduction across the 8 waves.
// ---------------------------------------------------------------------------
template <bool WBF16, bool WRITE_BF16, bool APPLY_ELU>
__global__ __launch_bounds__(256)
void moe_blend_layer(const __bf16* __restrict__ A,   // M x K activations (bf16)
                     const void*   __restrict__ Wv,  // E x N x K weights
                     const float*  __restrict__ EB,  // E x N biases (f32)
                     const float*  __restrict__ BC,  // M x E blend coeffs
                     void* __restrict__ outp,        // M x N (bf16 or f32)
                     int K, int N)
{
    const int lane   = threadIdx.x & 31;
    const int e      = threadIdx.x >> 5;   // expert = wave id
    const int lanelo = lane & 15;
    const int lanehi = lane >> 4;

    const int nbase = blockIdx.x * 16;
    const int mbase = blockIdx.y * 16;

    // A: lane supplies row (mbase+lanelo); K-run [k0+16*lanehi,+16) contiguous.
    const __bf16* __restrict__ arow =
        A + (size_t)(mbase + lanelo) * K + 16 * lanehi;
    // B: lane supplies column n = nbase+lanelo (= row n of W_e), K-run
    // [k0 + 16*lanehi, +16) contiguous in the row-major weight.
    const size_t woff = (size_t)e * N * K + (size_t)(nbase + lanelo) * K
                      + 16 * lanehi;

    v8f acc0 = (v8f){};
    v8f acc1 = (v8f){};

#pragma unroll 4
    for (int k0 = 0; k0 < K; k0 += 32) {
        v8bf alo = *(const v8bf*)(arow + k0);
        v8bf ahi = *(const v8bf*)(arow + k0 + 8);
        v16bf a = __builtin_shufflevector(alo, ahi,
                                          0, 1, 2, 3, 4, 5, 6, 7,
                                          8, 9, 10, 11, 12, 13, 14, 15);
        v16bf b;
        if constexpr (WBF16) {
            const __bf16* wp = (const __bf16*)Wv + woff + k0;
            v8bf blo = *(const v8bf*)(wp);
            v8bf bhi = *(const v8bf*)(wp + 8);
            b = __builtin_shufflevector(blo, bhi,
                                        0, 1, 2, 3, 4, 5, 6, 7,
                                        8, 9, 10, 11, 12, 13, 14, 15);
        } else {
            const float* wp = (const float*)Wv + woff + k0;
            f32x4 w0 = *(const f32x4*)(wp + 0);
            f32x4 w1 = *(const f32x4*)(wp + 4);
            f32x4 w2 = *(const f32x4*)(wp + 8);
            f32x4 w3 = *(const f32x4*)(wp + 12);
#pragma unroll
            for (int i = 0; i < 4; ++i) {
                b[i]      = (__bf16)w0[i];
                b[4 + i]  = (__bf16)w1[i];
                b[8 + i]  = (__bf16)w2[i];
                b[12 + i] = (__bf16)w3[i];
            }
        }
        // Alternate accumulators so consecutive WMMAs are independent.
        if ((k0 >> 5) & 1)
            acc1 = __builtin_amdgcn_wmma_f32_16x16x32_bf16(
                false, a, false, b, (short)0, acc1, false, false);
        else
            acc0 = __builtin_amdgcn_wmma_f32_16x16x32_bf16(
                false, a, false, b, (short)0, acc0, false, false);
    }

    // ---- Cross-wave expert blend via LDS ----
    // C/D layout: VGPR r -> row mbase + r + 8*lanehi, col nbase + lanelo.
    __shared__ float red[MOE_NEXP][256];   // 8 KB
#pragma unroll
    for (int r = 0; r < 8; ++r) {
        const int m = mbase + r + 8 * lanehi;
        const float bs = BC[m * MOE_NEXP + e];
        red[e][(r + 8 * lanehi) * 16 + lanelo] = bs * (acc0[r] + acc1[r]);
    }
    __syncthreads();

    // Thread t owns output element (m = t>>4, n = t&15) of the tile.
    const int tm = threadIdx.x >> 4;
    const int tn = threadIdx.x & 15;
    const int m  = mbase + tm;
    const int n  = nbase + tn;

    f32x4 bc0 = *(const f32x4*)(BC + m * MOE_NEXP);
    f32x4 bc1 = *(const f32x4*)(BC + m * MOE_NEXP + 4);
    float v = 0.0f;
#pragma unroll
    for (int q = 0; q < 4; ++q) {
        v += red[q][threadIdx.x]     + bc0[q] * EB[(size_t)q * N + n];
        v += red[4 + q][threadIdx.x] + bc1[q] * EB[(size_t)(4 + q) * N + n];
    }
    if (APPLY_ELU) v = v > 0.0f ? v : (__expf(v) - 1.0f);
    if (WRITE_BF16)
        ((__bf16*)outp)[(size_t)m * N + n] = (__bf16)v;
    else
        ((float*)outp)[(size_t)m * N + n] = v;
}

extern "C" void kernel_launch(void* const* d_in, const int* in_sizes, int n_in,
                              void* d_out, int out_size, void* d_ws, size_t ws_size,
                              hipStream_t stream) {
    (void)in_sizes; (void)n_in; (void)out_size;

    const float* x   = (const float*)d_in[0];  // (128, 512)
    const float* BC  = (const float*)d_in[1];  // (128, 8)
    const float* EW0 = (const float*)d_in[2];  // (8, 1024, 512)
    const float* EB0 = (const float*)d_in[3];  // (8, 1024)
    const float* EW1 = (const float*)d_in[4];  // (8, 1024, 1024)
    const float* EB1 = (const float*)d_in[5];  // (8, 1024)
    const float* EW2 = (const float*)d_in[6];  // (8, 512, 1024)
    const float* EB2 = (const float*)d_in[7];  // (8, 512)
    float* out = (float*)d_out;                // (128, 512)

    const int nx  = MOE_BATCH * 512;           // 65536
    const int nw0 = MOE_NEXP * 1024 * 512;     // 4,194,304
    const int nw1 = MOE_NEXP * 1024 * 1024;    // 8,388,608
    const int nw2 = MOE_NEXP * 512 * 1024;     // 4,194,304

    // Workspace layout: bf16 activations, then (optionally) bf16 weights.
    __bf16* xb  = (__bf16*)d_ws;
    __bf16* h0b = xb + nx;
    __bf16* h1b = h0b + (size_t)MOE_BATCH * 1024;
    __bf16* w0b = h1b + (size_t)MOE_BATCH * 1024;
    __bf16* w1b = w0b + nw0;
    __bf16* w2b = w1b + nw1;

    const size_t act_bytes = ((size_t)nx + 2u * MOE_BATCH * 1024) * sizeof(__bf16);
    const size_t need      = act_bytes + ((size_t)nw0 + nw1 + nw2) * sizeof(__bf16);
    const bool   pre       = ws_size >= need;   // constant per process: deterministic

    dim3 block(256);

    // Layer-0 input -> bf16 (always).
    f32_to_bf16_kernel<<<dim3(nx / 2048), block, 0, stream>>>(x, xb, nx);

    if (pre) {
        // Pre-convert all expert weights to bf16 (L2-resident on replay).
        f32_to_bf16_kernel<<<dim3(nw0 / 2048), block, 0, stream>>>(EW0, w0b, nw0);
        f32_to_bf16_kernel<<<dim3(nw1 / 2048), block, 0, stream>>>(EW1, w1b, nw1);
        f32_to_bf16_kernel<<<dim3(nw2 / 2048), block, 0, stream>>>(EW2, w2b, nw2);

        moe_blend_layer<true, true, true><<<dim3(64, 8), block, 0, stream>>>(
            xb, w0b, EB0, BC, h0b, 512, 1024);
        moe_blend_layer<true, true, true><<<dim3(64, 8), block, 0, stream>>>(
            h0b, w1b, EB1, BC, h1b, 1024, 1024);
        moe_blend_layer<true, false, false><<<dim3(32, 8), block, 0, stream>>>(
            h1b, w2b, EB2, BC, out, 1024, 512);
    } else {
        // Fallback: convert weights in-register inside the GEMM.
        moe_blend_layer<false, true, true><<<dim3(64, 8), block, 0, stream>>>(
            xb, EW0, EB0, BC, h0b, 512, 1024);
        moe_blend_layer<false, true, true><<<dim3(64, 8), block, 0, stream>>>(
            h0b, EW1, EB1, BC, h1b, 1024, 1024);
        moe_blend_layer<false, false, false><<<dim3(32, 8), block, 0, stream>>>(
            h1b, EW2, EB2, BC, out, 1024, 512);
    }
}
